// TorchLMHeadCE_49795850830126
// MI455X (gfx1250) — matrix operand order repaired and checked by reading, not compiled, not run on hardware
//
#include <hip/hip_runtime.h>
#include <math.h>

#define H_DIM   2048
#define V_DIM   50257
#define N_TOK   4096
#define VTILE   512                              // cols per block (8 waves * 4 N-tiles)
#define NUM_VT  ((V_DIM + VTILE - 1) / VTILE)    // 99
#define MTILE   32                               // rows per block (2 M-tiles per wave)
#define NUM_MT  (N_TOK / MTILE)                  // 128
#define KSTAGE  512
#define A_STRIDE 520                             // padded halves/row (bank-conflict free)
#define SOFTCAPF 30.0f
#define EPSF     0.1f
#define ZSCALE   1e-4f

typedef __attribute__((ext_vector_type(16))) __bf16 v16bf;
typedef __attribute__((ext_vector_type(8)))  float  v8f;
typedef __attribute__((ext_vector_type(4)))  unsigned int v4u;
typedef __attribute__((ext_vector_type(8)))  int v8i;
typedef __attribute__((ext_vector_type(4)))  int v4i;

union BFrag { uint4 q[2]; v16bf v; };

#if __has_builtin(__builtin_amdgcn_tensor_load_to_lds) && __has_builtin(__builtin_amdgcn_s_wait_tensorcnt)
#define USE_TDM 1
#else
#define USE_TDM 0
#endif

// -------- Kernel 0: fp32 -> bf16 bulk convert (count divisible by 4) --------
__global__ __launch_bounds__(256)
void convert_f32_bf16(const float* __restrict__ src, __bf16* __restrict__ dst, size_t count)
{
    size_t i = ((size_t)blockIdx.x * 256 + threadIdx.x) * 4;
    if (i < count) {
        float4 f = *(const float4*)(src + i);
        union { __bf16 h[4]; uint2 u; } p;
        p.h[0] = (__bf16)f.x; p.h[1] = (__bf16)f.y;
        p.h[2] = (__bf16)f.z; p.h[3] = (__bf16)f.w;
        *(uint2*)(dst + i) = p.u;
    }
}

// -------- Kernel 1: bf16-WMMA GEMM (TDM-staged A) + softcap + chunk stats --------
template<bool WBF16>
__global__ __launch_bounds__(256)
void lmhead_partial(const __bf16* __restrict__ xb,
                    const long long* __restrict__ y,
                    const float* __restrict__ Wf,
                    const __bf16* __restrict__ Wb,
                    float* __restrict__ wsMax,
                    float* __restrict__ wsSum,
                    float* __restrict__ wsLog,
                    float* __restrict__ wsTgt)
{
    extern __shared__ char smem[];
    __bf16* As     = (__bf16*)smem;                               // MTILE * A_STRIDE bf16
    float*  redMax = (float*)(smem + MTILE * A_STRIDE * 2);       // 8 waves * 32 rows
    float*  redSum = redMax + 8 * MTILE;
    float*  redLog = redSum + 8 * MTILE;
    int*    yTile  = (int*)(redLog + 8 * MTILE);                  // 32 labels

    const int tid  = threadIdx.x;
    const int wave = tid >> 5;
    const int lane = tid & 31;
    const int hi   = lane >> 4;
    const int l15  = lane & 15;

    const int rowBase = blockIdx.x * MTILE;
    const int vBase   = blockIdx.y * VTILE + wave * 64;           // 4 N-tiles per wave

    if (tid < MTILE) yTile[tid] = (int)y[rowBase + tid];

    int col[4];
    const float*  wptrF[4];
    const __bf16* wptrB[4];
    #pragma unroll
    for (int n = 0; n < 4; ++n) {
        col[n] = vBase + n * 16 + l15;
        int cc = col[n] < V_DIM ? col[n] : (V_DIM - 1);
        if constexpr (WBF16) wptrB[n] = Wb + (size_t)cc * H_DIM + hi * 16;
        else                 wptrF[n] = Wf + (size_t)cc * H_DIM + hi * 16;
    }

    v8f acc[2][4];
    #pragma unroll
    for (int m = 0; m < 2; ++m)
        #pragma unroll
        for (int n = 0; n < 4; ++n)
            #pragma unroll
            for (int v = 0; v < 8; ++v) acc[m][n][v] = 0.0f;

    for (int kk = 0; kk < H_DIM; kk += KSTAGE) {
        // ---- Stage x tile (already bf16) into padded LDS ----
#if USE_TDM
        if (wave == 0) {
            // TDM D#: 2D tile KSTAGE x MTILE bf16, row stride H, LDS pad 16B per row
            unsigned long long ga = (unsigned long long)(uintptr_t)
                (xb + (size_t)rowBase * H_DIM + kk);
            unsigned int ldsOff = (unsigned int)(uintptr_t)As;    // low 32 bits = LDS byte offset
            v4u g0 = { 1u,                                        // count=1, user mode
                       ldsOff,
                       (unsigned int)ga,
                       (unsigned int)((ga >> 32) & 0x01FFFFFFu) | 0x80000000u }; // type=2
            v8i g1 = { (int)((1u << 16)      // data_size = 1 (2 bytes)
                           | (1u << 20)      // pad_enable
                           | (7u << 22)      // pad_interval: 256 DWORDs (one row)
                           | (3u << 25)),    // pad_amount: 4 DWORDs = 16 B
                       (int)((KSTAGE & 0xFFFF) << 16),            // tensor_dim0[15:0]
                       (int)((MTILE  & 0xFFFF) << 16),            // dim0 hi | tensor_dim1 lo
                       (int)((KSTAGE & 0xFFFF) << 16),            // dim1 hi | tile_dim0
                       (int)MTILE,                                // tile_dim1 | tile_dim2=0
                       (int)H_DIM,                                // tensor_dim0_stride lo
                       0, 0 };
            v4i gz4 = { 0, 0, 0, 0 };
            v8i gz8 = { 0, 0, 0, 0, 0, 0, 0, 0 };
            __builtin_amdgcn_tensor_load_to_lds(g0, g1, gz4, gz4, gz8, 0);
            __builtin_amdgcn_s_wait_tensorcnt(0);
        }
#else
        for (int i = tid; i < MTILE * KSTAGE / 8; i += 256) {
            int flat = i * 8;
            int r = flat >> 9;               // / KSTAGE
            int k = flat & (KSTAGE - 1);
            uint4 q = *(const uint4*)(xb + (size_t)(rowBase + r) * H_DIM + kk + k);
            *(uint4*)(As + r * A_STRIDE + k) = q;
        }
#endif
        __syncthreads();

        for (int k0 = 0; k0 < KSTAGE; k0 += 32) {
            // A fragments (16-bit A 16x32 layout), one per M-tile
            BFrag a[2];
            #pragma unroll
            for (int m = 0; m < 2; ++m) {
                const __bf16* ap = As + (m * 16 + l15) * A_STRIDE + k0 + hi * 16;
                a[m].q[0] = *(const uint4*)(ap);
                a[m].q[1] = *(const uint4*)(ap + 8);
            }
            #pragma unroll
            for (int n = 0; n < 4; ++n) {
                v16bf b;
                if constexpr (WBF16) {
                    // 16 contiguous bf16 along K: two b128 loads, no cvt
                    const __bf16* wp = wptrB[n] + kk + k0;
                    BFrag bu;
                    bu.q[0] = *(const uint4*)(wp);
                    bu.q[1] = *(const uint4*)(wp + 8);
                    __builtin_prefetch((const void*)(wp + 32), 0, 3);
                    b = bu.v;
                } else {
                    const float* wp = wptrF[n] + kk + k0;  // 16 contiguous fp32 along K
                    float4 f0 = *(const float4*)(wp + 0);
                    float4 f1 = *(const float4*)(wp + 4);
                    float4 f2 = *(const float4*)(wp + 8);
                    float4 f3 = *(const float4*)(wp + 12);
                    __builtin_prefetch((const void*)(wp + 32), 0, 3);
                    b[0]=(__bf16)f0.x;  b[1]=(__bf16)f0.y;  b[2]=(__bf16)f0.z;  b[3]=(__bf16)f0.w;
                    b[4]=(__bf16)f1.x;  b[5]=(__bf16)f1.y;  b[6]=(__bf16)f1.z;  b[7]=(__bf16)f1.w;
                    b[8]=(__bf16)f2.x;  b[9]=(__bf16)f2.y;  b[10]=(__bf16)f2.z; b[11]=(__bf16)f2.w;
                    b[12]=(__bf16)f3.x; b[13]=(__bf16)f3.y; b[14]=(__bf16)f3.z; b[15]=(__bf16)f3.w;
                }
                #pragma unroll
                for (int m = 0; m < 2; ++m)            // reuse B across 2 M-tiles
                    acc[m][n] = __builtin_amdgcn_wmma_f32_16x16x32_bf16(
                        false, a[m].v, false, b, (short)0, acc[m][n], false, false);
            }
        }
        __syncthreads();
    }

    // ---- Softcap in place: l = 30*tanh(l/30) ----
    const float invCap = 1.0f / SOFTCAPF;
    #pragma unroll
    for (int m = 0; m < 2; ++m)
        #pragma unroll
        for (int n = 0; n < 4; ++n)
            #pragma unroll
            for (int v = 0; v < 8; ++v)
                acc[m][n][v] = SOFTCAPF * tanhf(acc[m][n][v] * invCap);

    bool valid[4];
    #pragma unroll
    for (int n = 0; n < 4; ++n) valid[n] = col[n] < V_DIM;

    // C layout: VGPR v, half-wave hi -> row M = m*16 + hi*8 + v; N = l15 in tile n
    float rmax[2][8], rsum[2][8], rlog[2][8];
    #pragma unroll
    for (int m = 0; m < 2; ++m)
        #pragma unroll
        for (int v = 0; v < 8; ++v) {
            float mx = -INFINITY;
            #pragma unroll
            for (int n = 0; n < 4; ++n) if (valid[n]) mx = fmaxf(mx, acc[m][n][v]);
            rmax[m][v] = mx;
        }
    #pragma unroll
    for (int msk = 1; msk <= 8; msk <<= 1)
        #pragma unroll
        for (int m = 0; m < 2; ++m)
            #pragma unroll
            for (int v = 0; v < 8; ++v)
                rmax[m][v] = fmaxf(rmax[m][v], __shfl_xor(rmax[m][v], msk, 32));

    #pragma unroll
    for (int m = 0; m < 2; ++m)
        #pragma unroll
        for (int v = 0; v < 8; ++v) {
            int r = m * 16 + hi * 8 + v;
            int yv = yTile[r];
            float s = 0.0f, L = 0.0f;
            #pragma unroll
            for (int n = 0; n < 4; ++n) {
                if (valid[n]) {
                    float l = acc[m][n][v];
                    s += __expf(l - rmax[m][v]);
                    L += l;
                    if (col[n] == yv) wsTgt[rowBase + r] = l;   // unique owner lane
                }
            }
            rsum[m][v] = s; rlog[m][v] = L;
        }
    #pragma unroll
    for (int msk = 1; msk <= 8; msk <<= 1)
        #pragma unroll
        for (int m = 0; m < 2; ++m)
            #pragma unroll
            for (int v = 0; v < 8; ++v) {
                rsum[m][v] += __shfl_xor(rsum[m][v], msk, 32);
                rlog[m][v] += __shfl_xor(rlog[m][v], msk, 32);
            }

    if (l15 == 0) {
        #pragma unroll
        for (int m = 0; m < 2; ++m)
            #pragma unroll
            for (int v = 0; v < 8; ++v) {
                int r = m * 16 + hi * 8 + v;
                redMax[wave * MTILE + r] = rmax[m][v];
                redSum[wave * MTILE + r] = rsum[m][v];
                redLog[wave * MTILE + r] = rlog[m][v];
            }
    }
    __syncthreads();

    if (tid < MTILE) {
        float M = -INFINITY;
        #pragma unroll
        for (int w = 0; w < 8; ++w) M = fmaxf(M, redMax[w * MTILE + tid]);
        float S = 0.0f, L = 0.0f;
        #pragma unroll
        for (int w = 0; w < 8; ++w) {
            S += redSum[w * MTILE + tid] * __expf(redMax[w * MTILE + tid] - M);
            L += redLog[w * MTILE + tid];
        }
        size_t o = (size_t)blockIdx.y * N_TOK + rowBase + tid;
        wsMax[o] = M; wsSum[o] = S; wsLog[o] = L;
    }
}

// -------- Kernel 2: merge V-chunks per row, CE + z-loss, block-reduce --------
__global__ __launch_bounds__(256)
void lmhead_rows(const float* __restrict__ wsMax, const float* __restrict__ wsSum,
                 const float* __restrict__ wsLog, const float* __restrict__ wsTgt,
                 const long long* __restrict__ y, float* __restrict__ blockOut)
{
    __shared__ float sv[256], sc[256];
    const int tid = threadIdx.x;
    const int row = blockIdx.x * 256 + tid;
    float val = 0.0f, cnt = 0.0f;
    if (row < N_TOK) {
        long long yv = y[row];
        if (yv != -100) {
            float M = -INFINITY;
            for (int c = 0; c < NUM_VT; ++c)
                M = fmaxf(M, wsMax[(size_t)c * N_TOK + row]);
            float S = 0.0f, L = 0.0f;
            for (int c = 0; c < NUM_VT; ++c) {
                S += wsSum[(size_t)c * N_TOK + row] * __expf(wsMax[(size_t)c * N_TOK + row] - M);
                L += wsLog[(size_t)c * N_TOK + row];
            }
            float lse = M + logf(S);
            float ce  = lse - (1.0f - EPSF) * wsTgt[row] - EPSF * (L / (float)V_DIM);
            val = ce + ZSCALE * lse * lse;
            cnt = 1.0f;
        }
    }
    sv[tid] = val; sc[tid] = cnt;
    __syncthreads();
    for (int s = 128; s > 0; s >>= 1) {
        if (tid < s) { sv[tid] += sv[tid + s]; sc[tid] += sc[tid + s]; }
        __syncthreads();
    }
    if (tid == 0) { blockOut[2 * blockIdx.x] = sv[0]; blockOut[2 * blockIdx.x + 1] = sc[0]; }
}

// -------- Kernel 3: deterministic final sum --------
__global__ void lmhead_final(const float* __restrict__ blockOut, float* __restrict__ out, int nblk)
{
    if (threadIdx.x == 0 && blockIdx.x == 0) {
        float s = 0.0f, c = 0.0f;
        for (int i = 0; i < nblk; ++i) { s += blockOut[2 * i]; c += blockOut[2 * i + 1]; }
        out[0] = s / c;
    }
}

extern "C" void kernel_launch(void* const* d_in, const int* in_sizes, int n_in,
                              void* d_out, int out_size, void* d_ws, size_t ws_size,
                              hipStream_t stream) {
    const float*     x = (const float*)d_in[0];
    const long long* y = (const long long*)d_in[1];
    const float*     W = (const float*)d_in[2];

    const size_t PER        = (size_t)NUM_VT * N_TOK;             // 99 * 4096
    const size_t floatCnt   = 3 * PER + N_TOK + 64;
    const size_t floatBytes = floatCnt * 4;                       // 16B-aligned (checked)
    const size_t xbBytes    = (size_t)N_TOK * H_DIM * 2;          // 16 MB
    const size_t wbBytes    = (size_t)V_DIM * H_DIM * 2;          // ~206 MB

    float*  ws       = (float*)d_ws;
    float*  wsMax    = ws;
    float*  wsSum    = ws + PER;
    float*  wsLog    = ws + 2 * PER;
    float*  wsTgt    = ws + 3 * PER;                              // N_TOK
    float*  blockOut = wsTgt + N_TOK;                             // 2 * 16
    __bf16* xb       = (__bf16*)((char*)d_ws + floatBytes);
    __bf16* wb       = (__bf16*)((char*)d_ws + floatBytes + xbBytes);

    const bool useWbf16 = ws_size >= floatBytes + xbBytes + wbBytes;

    // Kernel 0: x -> bf16 (and W -> bf16 when workspace allows)
    const size_t xCount = (size_t)N_TOK * H_DIM;
    convert_f32_bf16<<<dim3((unsigned)(xCount / 4 / 256)), 256, 0, stream>>>(x, xb, xCount);
    if (useWbf16) {
        const size_t wCount = (size_t)V_DIM * H_DIM;              // divisible by 1024
        convert_f32_bf16<<<dim3((unsigned)(wCount / 4 / 256)), 256, 0, stream>>>(W, wb, wCount);
    }

    // Kernel 1: fused GEMM + stats. Row-tile fastest -> L2 reuse of W chunk.
    dim3 gridA(NUM_MT, NUM_VT);                                   // 128 x 99
    size_t smemA = (size_t)MTILE * A_STRIDE * 2 + (3 * 8 * MTILE + MTILE) * 4;  // ~36.5 KB
    if (useWbf16)
        lmhead_partial<true><<<gridA, 256, smemA, stream>>>(xb, y, W, wb,
                                                            wsMax, wsSum, wsLog, wsTgt);
    else
        lmhead_partial<false><<<gridA, 256, smemA, stream>>>(xb, y, W, wb,
                                                             wsMax, wsSum, wsLog, wsTgt);

    const int nblk = N_TOK / 256;                                 // 16
    lmhead_rows<<<dim3(nblk), 256, 0, stream>>>(wsMax, wsSum, wsLog, wsTgt, y, blockOut);
    lmhead_final<<<1, 32, 0, stream>>>(blockOut, (float*)d_out, nblk);
}
